// SparseAttention3d_15152644620571
// MI455X (gfx1250) — matrix operand order repaired and checked by reading, not compiled, not run on hardware
//
#include <hip/hip_runtime.h>
#include <hip/hip_bf16.h>
#include <hip/hip_fp16.h>

#define NV   100000
#define NQ   20000
#define KNN  48
#define CH   128
#define FFD  256
#define OUTD 128
#define NH   8
#define DH   16
#define EPSF 1e-5f

typedef __attribute__((ext_vector_type(16))) _Float16 v16h;
typedef __attribute__((ext_vector_type(8)))  _Float16 v8h;
typedef __attribute__((ext_vector_type(8)))  float    v8f;
typedef unsigned int u32x4 __attribute__((ext_vector_type(4)));
typedef int          i32x8 __attribute__((ext_vector_type(8)));
typedef int          i32x4 __attribute__((ext_vector_type(4)));

// ---------------------------------------------------------------------------
// 16x16 output tile of  D += A(16 x kdim, LDS f16, row-major, lda halves)
//                         @ Wt(n0..n0+15 rows x kdim, f16 row-major [n][c])^T
// VGPR layouts used here (wave32; A/B k order is contiguous, not the ISA table's):
//   A: lane&15 = M;   lanes 0-15 hold K = 0..15, lanes 16-31 hold K = 16..31 (this code's order)
//   B: lane&15 = N;   lanes 0-15 hold K = 0..15, lanes 16-31 hold K = 16..31 (this code's order)
//   D: col = lane&15; rows = r + 8*(lane>=16)
// ---------------------------------------------------------------------------
__device__ __forceinline__ v8f wmma_tile(const _Float16* As, int lda,
                                         const _Float16* Wt, int ldw,
                                         int n0, int kdim, v8f acc) {
  const int lane = threadIdx.x & 31;
  const int m    = lane & 15;
  const int hi   = lane >> 4;                  // 0 or 1
  const _Float16* arow = As + m * lda + (hi << 4);        // +0 or +16 halves
  const _Float16* brow = Wt + (n0 + m) * ldw + (hi << 4);  // +0 or +16 halves
  for (int kk = 0; kk < kdim; kk += 32) {
    v8h a0 = *(const v8h*)(arow + kk);
    v8h a1 = *(const v8h*)(arow + kk + 8);
    v16h a;
    v16h b = *(const v16h*)(brow + kk);
#pragma unroll
    for (int i = 0; i < 8; i++) { a[i] = a0[i]; a[i + 8] = a1[i]; }
    acc = __builtin_amdgcn_wmma_f32_16x16x32_f16(false, a, false, b,
                                                 (short)0, acc, false, false);
  }
  return acc;
}

// ---------------------------------------------------------------------------
// Tensor Data Mover: DMA a 2-D f16 tile (rows x row_halves, contiguous) from
// global memory into LDS.  Descriptor per CDNA5 ISA ch.8 (D# groups 0/1).
// This toolchain exposes the 6-arg builtin (extra i32x8 group + cpol).
// ---------------------------------------------------------------------------
__device__ __forceinline__ unsigned lds_offset_of(const void* p) {
  return (unsigned)(unsigned long long)
         (__attribute__((address_space(3))) const char*)p;
}

__device__ __forceinline__ void tdm_load_2d_f16(unsigned lds_off,
                                                const void* gptr,
                                                unsigned rows,
                                                unsigned row_halves) {
  unsigned long long ga = (unsigned long long)gptr;
  u32x4 g0;
  g0[0] = 1u;                                   // count=1, user mode, no gather
  g0[1] = lds_off;                              // lds_addr  (bits 63:32)
  g0[2] = (unsigned)(ga & 0xffffffffu);         // global_addr[31:0] (bits 95:64)
  g0[3] = (unsigned)((ga >> 32) & 0x01ffffffu)  // global_addr[56:32]
        | (2u << 30);                           // type=2 ("image")
  i32x8 g1;
  g1[0] = (int)(1u << 16);                      // wg_mask=0, data_size=1 (2B)
  g1[1] = (int)((row_halves & 0xffffu) << 16);  // tensor_dim0[15:0] @ bits 63:48
  g1[2] = (int)((row_halves >> 16) & 0xffffu)   // tensor_dim0[31:16]
        | (int)((rows & 0xffffu) << 16);        // tensor_dim1[15:0]
  g1[3] = (int)((rows >> 16) & 0xffffu)         // tensor_dim1[31:16]
        | (int)((row_halves & 0xffffu) << 16);  // tile_dim0
  g1[4] = (int)(rows & 0xffffu);                // tile_dim1 (tile_dim2 = 0)
  g1[5] = (int)row_halves;                      // tensor_dim0_stride[31:0]
  g1[6] = 0;                                    // stride hi / dim1_stride lo
  g1[7] = 0;                                    // dim1_stride hi
  i32x4 z4 = {0, 0, 0, 0};
  i32x8 z8 = {0, 0, 0, 0, 0, 0, 0, 0};
  __builtin_amdgcn_tensor_load_to_lds(g0, g1, z4, z4, z8, 0);
}

// ---------------------------------------------------------------------------
// prep: f16 weight conversion (+ transposes) and BN accumulator zeroing
// ---------------------------------------------------------------------------
__global__ void prep_kernel(const float* __restrict__ inpw,
                            const float* __restrict__ out_w,
                            const float* __restrict__ w1,
                            const float* __restrict__ w2,
                            const float* __restrict__ w_out,
                            _Float16* wq_h, _Float16* wk_h, _Float16* wv_h,
                            _Float16* outw_h, _Float16* w1t, _Float16* w2t,
                            _Float16* woutt, float* stats) {
  int i = blockIdx.x * blockDim.x + threadIdx.x;
  if (i < CH * CH) {                 // direct copies: already [n][c] w.r.t. x@W^T
    wq_h[i]   = (_Float16)inpw[i];
    wk_h[i]   = (_Float16)inpw[CH * CH + i];
    wv_h[i]   = (_Float16)inpw[2 * CH * CH + i];
    outw_h[i] = (_Float16)out_w[i];
    int n = i >> 7, c = i & 127;     // w_out (C,OUT) used as x@w_out -> transpose
    woutt[i] = (_Float16)w_out[c * OUTD + n];
  }
  if (i < FFD * CH) {
    { int n = i >> 7, c = i & 127;   w1t[i] = (_Float16)w1[c * FFD + n]; }
    { int n = i >> 8, c = i & 255;   w2t[i] = (_Float16)w2[c * CH + n]; }
  }
  if (i < 8 * 128) stats[i] = 0.f;
}

// ---------------------------------------------------------------------------
// qproj: q = (relu(qc @ wq3 + bq3) @ wq^T + bq) * 1/sqrt(Dh)   16 queries/block
// ---------------------------------------------------------------------------
__global__ __launch_bounds__(128) void qproj_kernel(
    const float* __restrict__ qcoord, const float* __restrict__ wq3,
    const float* __restrict__ bq3, const _Float16* __restrict__ wq_h,
    const float* __restrict__ inpb, float* __restrict__ qbuf) {
  __shared__ _Float16 sQF[16 * CH];
  const int q0 = blockIdx.x * 16;
  const int tid = threadIdx.x;
  for (int e = tid; e < 16 * CH; e += 128) {
    int r = e >> 7, c = e & 127;
    int q = q0 + r;
    float x = qcoord[q * 3] * wq3[c] + qcoord[q * 3 + 1] * wq3[CH + c] +
              qcoord[q * 3 + 2] * wq3[2 * CH + c] + bq3[c];
    sQF[e] = (_Float16)(x > 0.f ? x : 0.f);
  }
  __syncthreads();
  const int lane = tid & 31, wave = tid >> 5;
  const int col = lane & 15, rbase = (lane >> 4) << 3;
  const float scale = 0.25f;  // 1/sqrt(16)
  for (int nt = wave; nt < 8; nt += 4) {
    int n0 = nt * 16;
    v8f acc = {};
    acc = wmma_tile(sQF, CH, wq_h, CH, n0, CH, acc);
    float b = inpb[n0 + col];
#pragma unroll
    for (int r = 0; r < 8; r++)
      qbuf[(size_t)(q0 + rbase + r) * CH + n0 + col] = (acc[r] + b) * scale;
  }
}

// ---------------------------------------------------------------------------
// attn: TDM-stage wk/wv into LDS, gather keys + pos-proj, K/V projection via
// WMMA (B operand from LDS), softmax, attn@V.  One query per 128-thread block.
// ---------------------------------------------------------------------------
__global__ __launch_bounds__(128) void attn_kernel(
    const float* __restrict__ vfeat, const float* __restrict__ vcoord,
    const float* __restrict__ qcoord, const float* __restrict__ wk3,
    const float* __restrict__ bk3, const float* __restrict__ inpb,
    const int* __restrict__ kidx, const unsigned char* __restrict__ kmask,
    const _Float16* __restrict__ wk_h, const _Float16* __restrict__ wv_h,
    const float* __restrict__ qbuf, _Float16* __restrict__ obuf) {
  __shared__ _Float16 sWK[CH * CH];   // 32 KB, filled by TDM
  __shared__ _Float16 sWV[CH * CH];   // 32 KB, filled by TDM
  __shared__ _Float16 sA[KNN * CH];
  __shared__ _Float16 sK[KNN * CH];
  __shared__ _Float16 sV[KNN * CH];
  __shared__ float    sScore[NH * KNN];
  __shared__ float    sQ[CH];
  __shared__ float    sRC[KNN * 3];
  __shared__ int      sIdx[KNN];

  const int q = blockIdx.x;
  const int tid = threadIdx.x;
  const int wave = tid >> 5;

  // kick off the weight DMA first so it overlaps the gather phase
  if (wave == 0) {
    tdm_load_2d_f16(lds_offset_of(sWK), wk_h, CH, CH);
  } else if (wave == 1) {
    tdm_load_2d_f16(lds_offset_of(sWV), wv_h, CH, CH);
  }

  if (tid < KNN) {
    int idx = kidx[q * KNN + tid];
    sIdx[tid] = idx;
    __builtin_prefetch(vfeat + (size_t)idx * CH, 0, 3);   // global_prefetch
    sRC[tid * 3 + 0] = vcoord[idx * 3 + 0] - qcoord[q * 3 + 0];
    sRC[tid * 3 + 1] = vcoord[idx * 3 + 1] - qcoord[q * 3 + 1];
    sRC[tid * 3 + 2] = vcoord[idx * 3 + 2] - qcoord[q * 3 + 2];
  }
  sQ[tid] = qbuf[(size_t)q * CH + tid];
  __syncthreads();

  // gather: thread t owns channel c=t for all 48 keys (hoists wk3 column)
  {
    const int c = tid;
    const float wx = wk3[c], wy = wk3[CH + c], wz = wk3[2 * CH + c];
    const float bb = bk3[c];
    for (int j = 0; j < KNN; j++) {
      float p = sRC[j * 3] * wx + sRC[j * 3 + 1] * wy + sRC[j * 3 + 2] * wz + bb;
      p = p > 0.f ? p : 0.f;
      sA[j * CH + c] = (_Float16)(vfeat[(size_t)sIdx[j] * CH + c] + p);
    }
  }
  if (wave < 2) __builtin_amdgcn_s_wait_tensorcnt(0);  // issuing waves drain TDM
  __syncthreads();

  const int lane = tid & 31;
  const int col = lane & 15, rbase = (lane >> 4) << 3;
  // 24 K-tiles + 24 V-tiles; each wave does 6 of each (uniform trip count)
  for (int t = wave * 6; t < wave * 6 + 6; t++) {
    int m0 = (t >> 3) * 16, n0 = (t & 7) * 16;
    v8f accK = {}, accV = {};
    accK = wmma_tile(sA + m0 * CH, CH, sWK, CH, n0, CH, accK);
    accV = wmma_tile(sA + m0 * CH, CH, sWV, CH, n0, CH, accV);
    float bk_ = inpb[CH + n0 + col];
    float bv_ = inpb[2 * CH + n0 + col];
#pragma unroll
    for (int r = 0; r < 8; r++) {
      sK[(m0 + rbase + r) * CH + n0 + col] = (_Float16)(accK[r] + bk_);
      sV[(m0 + rbase + r) * CH + n0 + col] = (_Float16)(accV[r] + bv_);
    }
  }
  __syncthreads();

  // per-head scores (q already scaled)
  for (int p = tid; p < NH * KNN; p += 128) {
    int h = p / KNN, j = p - h * KNN;
    float s;
    if (kmask[(size_t)q * KNN + j]) {
      s = -1e9f;
    } else {
      s = 0.f;
      const _Float16* kr = sK + j * CH + h * DH;
      const float* qr = sQ + h * DH;
#pragma unroll
      for (int d = 0; d < DH; d++) s += qr[d] * (float)kr[d];
    }
    sScore[h * KNN + j] = s;
  }
  __syncthreads();
  if (tid < NH) {
    float mx = -1e30f;
    for (int j = 0; j < KNN; j++) mx = fmaxf(mx, sScore[tid * KNN + j]);
    float sum = 0.f;
    for (int j = 0; j < KNN; j++) {
      float e = __expf(sScore[tid * KNN + j] - mx);
      sScore[tid * KNN + j] = e;
      sum += e;
    }
    float inv = 1.f / sum;
    for (int j = 0; j < KNN; j++) sScore[tid * KNN + j] *= inv;
  }
  __syncthreads();
  {
    const int c = tid, h = c >> 4;
    float acc = 0.f;
    for (int j = 0; j < KNN; j++) acc += sScore[h * KNN + j] * (float)sV[j * CH + c];
    obuf[(size_t)q * CH + c] = (_Float16)acc;
  }
}

// ---------------------------------------------------------------------------
// ffn: x = o@out_w^T + b ; x += relu(x@w1+b1)@w2+b2 ; BN1 stats.  16 q/block
// ---------------------------------------------------------------------------
__global__ __launch_bounds__(128) void ffn_kernel(
    const _Float16* __restrict__ obuf, const _Float16* __restrict__ outw_h,
    const float* __restrict__ out_b, const _Float16* __restrict__ w1t,
    const float* __restrict__ b1, const _Float16* __restrict__ w2t,
    const float* __restrict__ b2, float* __restrict__ xbuf,
    float* __restrict__ sum1, float* __restrict__ sq1) {
  __shared__ _Float16 sO[16 * CH];
  __shared__ _Float16 sXh[16 * CH];
  __shared__ float    sXf[16 * CH];
  __shared__ _Float16 sT[16 * FFD];
  const int q0 = blockIdx.x * 16, tid = threadIdx.x;
  for (int e = tid; e < 16 * CH; e += 128) sO[e] = obuf[(size_t)q0 * CH + e];
  __syncthreads();
  const int lane = tid & 31, wave = tid >> 5;
  const int col = lane & 15, rbase = (lane >> 4) << 3;
  for (int nt = wave; nt < 8; nt += 4) {
    int n0 = nt * 16;
    v8f acc = {};
    acc = wmma_tile(sO, CH, outw_h, CH, n0, CH, acc);
    float b = out_b[n0 + col];
#pragma unroll
    for (int r = 0; r < 8; r++) {
      float x = acc[r] + b;
      sXf[(rbase + r) * CH + n0 + col] = x;
      sXh[(rbase + r) * CH + n0 + col] = (_Float16)x;
    }
  }
  __syncthreads();
  for (int nt = wave; nt < 16; nt += 4) {
    int n0 = nt * 16;
    v8f acc = {};
    acc = wmma_tile(sXh, CH, w1t, CH, n0, CH, acc);
    float b = b1[n0 + col];
#pragma unroll
    for (int r = 0; r < 8; r++) {
      float t = acc[r] + b;
      sT[(rbase + r) * FFD + n0 + col] = (_Float16)(t > 0.f ? t : 0.f);
    }
  }
  __syncthreads();
  for (int nt = wave; nt < 8; nt += 4) {
    int n0 = nt * 16;
    v8f acc = {};
    acc = wmma_tile(sT, FFD, w2t, FFD, n0, FFD, acc);
    float b = b2[n0 + col];
    float s = 0.f, ss = 0.f;
#pragma unroll
    for (int r = 0; r < 8; r++) {
      float x = sXf[(rbase + r) * CH + n0 + col] + acc[r] + b;
      xbuf[(size_t)(q0 + rbase + r) * CH + n0 + col] = x;
      s += x;
      ss += x * x;
    }
    atomicAdd(&sum1[n0 + col], s);
    atomicAdd(&sq1[n0 + col], ss);
  }
}

// BN fold: per-channel scale/shift from accumulated sums
__global__ void bnprep_kernel(const float* __restrict__ sum,
                              const float* __restrict__ sq,
                              const float* __restrict__ g,
                              const float* __restrict__ be,
                              float* sv, float* tv, float invn) {
  int c = threadIdx.x;
  float m = sum[c] * invn;
  float var = sq[c] * invn - m * m;
  float s = g[c] * rsqrtf(var + EPSF);
  sv[c] = s;
  tv[c] = be[c] - m * s;
}

// head: z = bn1(x)@w_out + b_out ; BN2 stats.  16 q/block
__global__ __launch_bounds__(128) void head_kernel(
    const float* __restrict__ xbuf, const float* __restrict__ s1,
    const float* __restrict__ t1, const _Float16* __restrict__ woutt,
    const float* __restrict__ b_out, float* __restrict__ zbuf,
    float* __restrict__ sum2, float* __restrict__ sq2) {
  __shared__ _Float16 sXn[16 * CH];
  const int q0 = blockIdx.x * 16, tid = threadIdx.x;
  for (int e = tid; e < 16 * CH; e += 128) {
    int c = e & 127;
    sXn[e] = (_Float16)(xbuf[(size_t)q0 * CH + e] * s1[c] + t1[c]);
  }
  __syncthreads();
  const int lane = tid & 31, wave = tid >> 5;
  const int col = lane & 15, rbase = (lane >> 4) << 3;
  for (int nt = wave; nt < 8; nt += 4) {
    int n0 = nt * 16;
    v8f acc = {};
    acc = wmma_tile(sXn, CH, woutt, CH, n0, CH, acc);
    float b = b_out[n0 + col];
    float s = 0.f, ss = 0.f;
#pragma unroll
    for (int r = 0; r < 8; r++) {
      float z = acc[r] + b;
      zbuf[(size_t)(q0 + rbase + r) * OUTD + n0 + col] = z;
      s += z;
      ss += z * z;
    }
    atomicAdd(&sum2[n0 + col], s);
    atomicAdd(&sq2[n0 + col], ss);
  }
}

// out: y = relu(bn2(z))
__global__ void out_kernel(const float* __restrict__ zbuf,
                           const float* __restrict__ s2,
                           const float* __restrict__ t2,
                           float* __restrict__ y, int n) {
  int i = blockIdx.x * blockDim.x + threadIdx.x;
  if (i < n) {
    int c = i & (OUTD - 1);
    float v = zbuf[i] * s2[c] + t2[c];
    y[i] = v > 0.f ? v : 0.f;
  }
}

// ---------------------------------------------------------------------------
extern "C" void kernel_launch(void* const* d_in, const int* in_sizes, int n_in,
                              void* d_out, int out_size, void* d_ws, size_t ws_size,
                              hipStream_t stream) {
  const float* vfeat  = (const float*)d_in[0];
  const float* vcoord = (const float*)d_in[1];
  const float* qcoord = (const float*)d_in[2];
  const float* wq3    = (const float*)d_in[3];
  const float* bq3    = (const float*)d_in[4];
  const float* wk3    = (const float*)d_in[5];
  const float* bk3    = (const float*)d_in[6];
  const float* inpw   = (const float*)d_in[7];
  const float* inpb   = (const float*)d_in[8];
  const float* out_w  = (const float*)d_in[9];
  const float* out_b  = (const float*)d_in[10];
  const float* w1     = (const float*)d_in[11];
  const float* b1     = (const float*)d_in[12];
  const float* w2     = (const float*)d_in[13];
  const float* b2     = (const float*)d_in[14];
  const float* g1     = (const float*)d_in[15];
  const float* be1    = (const float*)d_in[16];
  const float* w_out  = (const float*)d_in[17];
  const float* b_out  = (const float*)d_in[18];
  const float* g2     = (const float*)d_in[19];
  const float* be2    = (const float*)d_in[20];
  const int*   kidx   = (const int*)d_in[21];
  const unsigned char* kmask = (const unsigned char*)d_in[22];

  char* ws = (char*)d_ws;
  size_t off = 0;
  auto carve = [&](size_t bytes) -> char* {
    char* p = ws + off;
    off = (off + bytes + 255) & ~(size_t)255;
    return p;
  };
  _Float16* wq_h   = (_Float16*)carve(CH * CH * 2);
  _Float16* wk_h   = (_Float16*)carve(CH * CH * 2);
  _Float16* wv_h   = (_Float16*)carve(CH * CH * 2);
  _Float16* outw_h = (_Float16*)carve(CH * CH * 2);
  _Float16* w1t    = (_Float16*)carve(FFD * CH * 2);
  _Float16* w2t    = (_Float16*)carve(FFD * CH * 2);
  _Float16* woutt  = (_Float16*)carve(CH * OUTD * 2);
  float*    stats  = (float*)carve(8 * 128 * 4);
  float*    qbuf   = (float*)carve((size_t)NQ * CH * 4);
  _Float16* obuf   = (_Float16*)carve((size_t)NQ * CH * 2);
  float*    xbuf   = (float*)carve((size_t)NQ * CH * 4);
  float*    zbuf   = (float*)carve((size_t)NQ * OUTD * 4);

  float* sum1 = stats + 0;
  float* sq1  = stats + 128;
  float* s1   = stats + 256;
  float* t1   = stats + 384;
  float* sum2 = stats + 512;
  float* sq2  = stats + 640;
  float* s2   = stats + 768;
  float* t2   = stats + 896;

  const float invn = 1.0f / (float)NQ;

  prep_kernel<<<(FFD * CH + 255) / 256, 256, 0, stream>>>(
      inpw, out_w, w1, w2, w_out, wq_h, wk_h, wv_h, outw_h, w1t, w2t, woutt, stats);

  qproj_kernel<<<NQ / 16, 128, 0, stream>>>(qcoord, wq3, bq3, wq_h, inpb, qbuf);

  attn_kernel<<<NQ, 128, 0, stream>>>(vfeat, vcoord, qcoord, wk3, bk3, inpb,
                                      kidx, kmask, wk_h, wv_h, qbuf, obuf);

  ffn_kernel<<<NQ / 16, 128, 0, stream>>>(obuf, outw_h, out_b, w1t, b1, w2t, b2,
                                          xbuf, sum1, sq1);

  bnprep_kernel<<<1, 128, 0, stream>>>(sum1, sq1, g1, be1, s1, t1, invn);

  head_kernel<<<NQ / 16, 128, 0, stream>>>(xbuf, s1, t1, woutt, b_out, zbuf,
                                           sum2, sq2);

  bnprep_kernel<<<1, 128, 0, stream>>>(sum2, sq2, g2, be2, s2, t2, invn);

  const int nOut = NQ * OUTD;
  out_kernel<<<(nOut + 255) / 256, 256, 0, stream>>>(zbuf, s2, t2, (float*)d_out, nOut);
}
